// SelfAttention_3058016714978
// MI455X (gfx1250) — matrix operand order repaired and checked by reading, not compiled, not run on hardware
//
#include <hip/hip_runtime.h>
#include <hip/hip_bf16.h>

#define DEVFN __device__ __forceinline__

typedef __attribute__((ext_vector_type(16))) _Float16 v16h;
typedef __attribute__((ext_vector_type(8)))  _Float16 v8h;
typedef __attribute__((ext_vector_type(8)))  float    v8f;

constexpr int NB  = 2;
constexpr int NS  = 2048;
constexpr int NE  = 1024;
constexpr int NH  = 16;
constexpr int NDH = 64;
constexpr int NM  = NB * NS;  // 4096 rows

// softmax scale = 1/sqrt(d_k) with d_k = S = 2048 (faithful to reference)
constexpr float SM_SCALE = 0.022097086912079608f;
constexpr float NEG_BIG  = -1.0e20f * SM_SCALE;  // masked logit after scaling

DEVFN v8f wmma16(v16h a, v16h b, v8f c) {
  // D = A(16x32 f16) * B(32x16 f16) + C(16x16 f32)
  return __builtin_amdgcn_wmma_f32_16x16x32_f16(false, a, false, b, (short)0, c,
                                                false, false);
}

// ---- fragment loaders (this code's contiguous k order, wave32) ----------------
// A 16x32 f16: lane l holds row (l&15); halves 0..15 = K = (l>>4)*16 + 0..15.
DEVFN v16h load_a_f32(const float* __restrict__ base, int ld) {
  const int lane = threadIdx.x & 31;
  const float* p = base + (size_t)(lane & 15) * ld + (lane >> 4) * 16;
  v16h r;
#pragma unroll
  for (int i = 0; i < 8; ++i) r[i] = (_Float16)p[i];
#pragma unroll
  for (int i = 0; i < 8; ++i) r[8 + i] = (_Float16)p[8 + i];
  return r;
}

DEVFN v16h load_a_f16(const _Float16* __restrict__ base, int ld) {
  const int lane = threadIdx.x & 31;
  const _Float16* p = base + (size_t)(lane & 15) * ld + (lane >> 4) * 16;
  v8h lo = *(const v8h*)p;
  v8h hs = *(const v8h*)(p + 8);
  v16h r;
#pragma unroll
  for (int i = 0; i < 8; ++i) { r[i] = lo[i]; r[8 + i] = hs[i]; }
  return r;
}

// B 32x16 f16 (NT source: row n of source is contiguous along K):
// lane l holds column n=(l&15); halves 0..15 = K = (l>>4)*16 + 0..15.
DEVFN v16h load_b_f32(const float* __restrict__ base, int ld) {
  const int lane = threadIdx.x & 31;
  const float* p = base + (size_t)(lane & 15) * ld + (lane >> 4) * 16;
  v16h r;
#pragma unroll
  for (int i = 0; i < 16; ++i) r[i] = (_Float16)p[i];
  return r;
}

DEVFN v16h load_b_f16(const _Float16* __restrict__ base, int ld) {
  const int lane = threadIdx.x & 31;
  return *(const v16h*)(base + (size_t)(lane & 15) * ld + (lane >> 4) * 16);
}

// ---- C/D tile store: lane l -> col (l&15); VGPR i -> row i + 8*(l>>4) ------
// MODE 0: f16 row-major [4096, 1024]
// MODE 1: f16 transposed V layout (B, H, DH, S)
// MODE 2: f32 row-major + bias (final output)
template <int MODE>
DEVFN void store_tile(const v8f acc, int mt, int nt, _Float16* __restrict__ outh,
                      float* __restrict__ outf, const float* __restrict__ bias) {
  const int lane = threadIdx.x & 31;
  const int ln = lane & 15, hi = lane >> 4;
  const int col = nt + ln;
  float bv = 0.0f;
  if (MODE == 2) bv = bias[col];
#pragma unroll
  for (int i = 0; i < 8; ++i) {
    const int row = mt + i + hi * 8;
    if (MODE == 0) {
      outh[(size_t)row * NE + col] = (_Float16)acc[i];
    } else if (MODE == 1) {
      const int bb = row >> 11;          // row = b*S + s
      const int ss = row & (NS - 1);
      const int hh = col >> 6;           // col = h*DH + d
      const int dd = col & (NDH - 1);
      outh[(((size_t)bb * NH + hh) * NDH + dd) * NS + ss] = (_Float16)acc[i];
    } else {
      outf[(size_t)row * NE + col] = acc[i] + bv;
    }
  }
}

// ---- GEMM-NT: C[4096,1024] = A[4096,1024] * W[1024,1024]^T -----------------
// 256 threads = 8 waves; wave -> 32x32 sub-tile (2x2 WMMA); block -> 64x128.
template <int MODE, bool AF16>
__global__ void __launch_bounds__(256) gemm_nt(
    const float* __restrict__ Af, const _Float16* __restrict__ Ah,
    const float* __restrict__ W, _Float16* __restrict__ outh,
    float* __restrict__ outf, const float* __restrict__ bias) {
  const int wave = threadIdx.x >> 5;
  const int m0 = blockIdx.x * 64 + (wave >> 2) * 32;
  const int n0 = blockIdx.y * 128 + (wave & 3) * 32;

  v8f a00 = {}, a01 = {}, a10 = {}, a11 = {};
#pragma unroll 4
  for (int k = 0; k < NE; k += 32) {
    v16h fa0, fa1;
    if constexpr (AF16) {
      fa0 = load_a_f16(Ah + (size_t)m0 * NE + k, NE);
      fa1 = load_a_f16(Ah + (size_t)(m0 + 16) * NE + k, NE);
    } else {
      fa0 = load_a_f32(Af + (size_t)m0 * NE + k, NE);
      fa1 = load_a_f32(Af + (size_t)(m0 + 16) * NE + k, NE);
    }
    v16h fb0 = load_b_f32(W + (size_t)n0 * NE + k, NE);
    v16h fb1 = load_b_f32(W + (size_t)(n0 + 16) * NE + k, NE);
    a00 = wmma16(fa0, fb0, a00);
    a01 = wmma16(fa0, fb1, a01);
    a10 = wmma16(fa1, fb0, a10);
    a11 = wmma16(fa1, fb1, a11);
  }
  store_tile<MODE>(a00, m0, n0, outh, outf, bias);
  store_tile<MODE>(a01, m0, n0 + 16, outh, outf, bias);
  store_tile<MODE>(a10, m0 + 16, n0, outh, outf, bias);
  store_tile<MODE>(a11, m0 + 16, n0 + 16, outh, outf, bias);
}

// ---- Flash attention: grid (S/64, H, B), 128 threads = 4 waves -------------
// Wave owns 16 query rows; streams keys in blocks of 32 with online softmax.
__global__ void __launch_bounds__(128) attn_kernel(
    const _Float16* __restrict__ Qh,   // (B, S, H*DH)
    const _Float16* __restrict__ Kh,   // (B, S, H*DH)
    const _Float16* __restrict__ Vt,   // (B, H, DH, S)
    const int* __restrict__ mask,      // (B, S)
    _Float16* __restrict__ ctx) {      // (B, S, H*DH)
  __shared__ _Float16 pbuf[4][16 * 32];

  const int wave = threadIdx.x >> 5;
  const int lane = threadIdx.x & 31;
  const int ln = lane & 15, hi = lane >> 4;
  const int b = blockIdx.z, h = blockIdx.y;
  const int q0 = blockIdx.x * 64 + wave * 16;

  // Q fragments for d-steps {0..31} and {32..63}; reused for every key block
  const _Float16* Qb = Qh + ((size_t)b * NS + q0) * NE + h * NDH;
  const v16h qa0 = load_a_f16(Qb, NE);
  const v16h qa1 = load_a_f16(Qb + 32, NE);

  v8f o0 = {}, o1 = {}, o2 = {}, o3 = {};
  float mrow[8], lrow[8];
#pragma unroll
  for (int i = 0; i < 8; ++i) { mrow[i] = -3.0e38f; lrow[i] = 0.0f; }

  const int* mrowp = mask + (size_t)b * NS;
  _Float16* lp = pbuf[wave];

  for (int kb = 0; kb < NS; kb += 32) {
    // ---- scores: S(16x32) = Q(16x64) . K(32x64)^T, two 16x16 C tiles ------
    const _Float16* Kb = Kh + ((size_t)b * NS + kb) * NE + h * NDH;
    v8f s0 = {}, s1 = {};
    s0 = wmma16(qa0, load_b_f16(Kb, NE), s0);
    s0 = wmma16(qa1, load_b_f16(Kb + 32, NE), s0);
    s1 = wmma16(qa0, load_b_f16(Kb + (size_t)16 * NE, NE), s1);
    s1 = wmma16(qa1, load_b_f16(Kb + (size_t)16 * NE + 32, NE), s1);

    const bool z0 = (mrowp[kb + ln] == 0);
    const bool z1 = (mrowp[kb + 16 + ln] == 0);

    // ---- online softmax over this 32-key block ----------------------------
#pragma unroll
    for (int i = 0; i < 8; ++i) {
      float e0 = z0 ? NEG_BIG : s0[i] * SM_SCALE;
      float e1 = z1 ? NEG_BIG : s1[i] * SM_SCALE;
      float mx = fmaxf(e0, e1);
      mx = fmaxf(mx, __shfl_xor(mx, 1));
      mx = fmaxf(mx, __shfl_xor(mx, 2));
      mx = fmaxf(mx, __shfl_xor(mx, 4));
      mx = fmaxf(mx, __shfl_xor(mx, 8));
      const float nm = fmaxf(mrow[i], mx);
      const float c = __expf(mrow[i] - nm);
      mrow[i] = nm;
      const float p0 = __expf(e0 - nm);
      const float p1 = __expf(e1 - nm);
      float ps = p0 + p1;
      ps += __shfl_xor(ps, 1);
      ps += __shfl_xor(ps, 2);
      ps += __shfl_xor(ps, 4);
      ps += __shfl_xor(ps, 8);
      lrow[i] = lrow[i] * c + ps;
      o0[i] *= c; o1[i] *= c; o2[i] *= c; o3[i] *= c;
      // C-layout -> LDS (row-major 16x32) for re-read in A-layout
      lp[(i + hi * 8) * 32 + ln] = (_Float16)p0;
      lp[(i + hi * 8) * 32 + 16 + ln] = (_Float16)p1;
    }

    asm volatile("s_wait_dscnt 0" ::: "memory");  // RAW: LDS stores -> loads

    // P as A fragment (16 rows x 32 keys) from LDS
    const _Float16* ap = lp + ln * 32 + hi * 16;
    v8h plo = *(const v8h*)ap;
    v8h phi = *(const v8h*)(ap + 8);
    v16h pa;
#pragma unroll
    for (int i = 0; i < 8; ++i) { pa[i] = plo[i]; pa[8 + i] = phi[i]; }

    // ---- O += P(16x32) . V(32x64); V^T rows contiguous along keys ---------
    const _Float16* Vb = Vt + ((size_t)b * NH + h) * NDH * NS + kb;
    o0 = wmma16(pa, load_b_f16(Vb, NS), o0);
    o1 = wmma16(pa, load_b_f16(Vb + (size_t)16 * NS, NS), o1);
    o2 = wmma16(pa, load_b_f16(Vb + (size_t)32 * NS, NS), o2);
    o3 = wmma16(pa, load_b_f16(Vb + (size_t)48 * NS, NS), o3);
  }

  // ---- normalize and store context f16 (B, S, H*DH) ------------------------
  _Float16* cp = ctx + ((size_t)b * NS + q0) * NE + h * NDH;
#pragma unroll
  for (int i = 0; i < 8; ++i) {
    const float inv = 1.0f / lrow[i];
    const int row = i + hi * 8;
    _Float16* r = cp + (size_t)row * NE + ln;
    r[0]  = (_Float16)(o0[i] * inv);
    r[16] = (_Float16)(o1[i] * inv);
    r[32] = (_Float16)(o2[i] * inv);
    r[48] = (_Float16)(o3[i] * inv);
  }
}

extern "C" void kernel_launch(void* const* d_in, const int* in_sizes, int n_in,
                              void* d_out, int out_size, void* d_ws, size_t ws_size,
                              hipStream_t stream) {
  (void)in_sizes; (void)n_in; (void)out_size; (void)ws_size;
  const float* qin  = (const float*)d_in[0];
  const float* kin  = (const float*)d_in[1];
  const float* vin  = (const float*)d_in[2];
  const int*   mask = (const int*)d_in[3];
  const float* Wq   = (const float*)d_in[4];
  const float* Wk   = (const float*)d_in[5];
  const float* Wv   = (const float*)d_in[6];
  const float* Wo   = (const float*)d_in[7];
  const float* bo   = (const float*)d_in[8];
  float* out = (float*)d_out;

  _Float16* Qh  = (_Float16*)d_ws;                       // 8 MB each
  _Float16* Kh  = Qh + (size_t)NM * NE;
  _Float16* Vt  = Kh + (size_t)NM * NE;
  _Float16* ctx = Vt + (size_t)NM * NE;

  const dim3 gg(NM / 64, NE / 128);
  const dim3 gb(256);

  gemm_nt<0, false><<<gg, gb, 0, stream>>>(qin, nullptr, Wq, Qh, nullptr, nullptr);
  gemm_nt<0, false><<<gg, gb, 0, stream>>>(kin, nullptr, Wk, Kh, nullptr, nullptr);
  gemm_nt<1, false><<<gg, gb, 0, stream>>>(vin, nullptr, Wv, Vt, nullptr, nullptr);

  attn_kernel<<<dim3(NS / 64, NH, NB), dim3(128), 0, stream>>>(Qh, Kh, Vt, mask, ctx);

  gemm_nt<2, true><<<gg, gb, 0, stream>>>(nullptr, ctx, Wo, nullptr, out, bo);
}
